// KPFCNN_80564996539067
// MI455X (gfx1250) — matrix operand order repaired and checked by reading, not compiled, not run on hardware
//
#include <hip/hip_runtime.h>

// ---------------------------------------------------------------------------
// KPFCNN forward for MI455X (gfx1250, wave32).
// GEMM: v_wmma_f32_16x16x32_bf16. Block tile 64x64, 4 waves (128 thr),
// wave tile 32x32 -> 4 WMMA per K-step per wave. LDS double-buffered:
// one barrier per K-step, global loads overlap WMMA. B staged transposed
// so all fragment loads are ds_load_b128. Edge tiles: clamp+select.
// Gather/GN/maxpool stages are bandwidth-bound VALU kernels.
// ---------------------------------------------------------------------------

#define KPTS 15
#define HN 16
#define LRELU_NEG 0.1f
#define GN_EPS 1e-5f

typedef __bf16 bf16;
typedef bf16 v16bf __attribute__((ext_vector_type(16)));
typedef float v8f __attribute__((ext_vector_type(8)));

struct alignas(16) F4 { float x, y, z, w; };
struct alignas(8)  U2 { unsigned x, y; };
struct alignas(16) U4 { unsigned x, y, z, w; };
struct U8 { U4 lo, hi; };

__device__ __forceinline__ unsigned short f2bf(float f) {
  unsigned u = __builtin_bit_cast(unsigned, f);
  u += 0x7FFFu + ((u >> 16) & 1u);          // round-to-nearest-even
  return (unsigned short)(u >> 16);
}

// ---------------------------------------------------------------------------
// C[M,N] = A[M,Kd] * B[Kd,N] (+ bias). f32 in/out, bf16 WMMA compute.
// ---------------------------------------------------------------------------
__global__ __launch_bounds__(128) void k_gemm(
    const float* __restrict__ A, const float* __restrict__ B,
    const float* __restrict__ bias, float* __restrict__ C,
    int M, int Kd, int N) {
  __shared__ unsigned short As[2][64][40];  // [buf][m][k]
  __shared__ unsigned short Bt[2][64][40];  // [buf][n][k] (B transposed)
  int tid = threadIdx.x;
  int wave = tid >> 5, lane = tid & 31;
  int wm = wave >> 1, wn = wave & 1;        // 2x2 wave grid; wave tile 32x32
  int bm = blockIdx.y << 6;
  int bn = blockIdx.x << 6;
  int hh = lane >> 4;                       // lane half
  int ll = lane & 15;
  v8f acc00 = {0.f,0.f,0.f,0.f,0.f,0.f,0.f,0.f};
  v8f acc01 = acc00, acc10 = acc00, acc11 = acc00;

  int lrow = tid >> 3;                      // 0..15
  int lc4 = (tid & 7) << 2;                 // 0,4,..,28

  bool fullM = (bm + 64 <= M);
  bool fullN = (bn + 64 <= N);
  bool vecA = fullM && ((Kd & 3) == 0);
  bool vecB = fullN && ((N & 3) == 0);

  auto stage = [&](int b, int kk0) {
    bool fullK = (kk0 + 32 <= Kd);
    // A tile 64x32
    if (vecA && fullK) {
#pragma unroll
      for (int p = 0; p < 4; p++) {
        int r = lrow + (p << 4);
        F4 v = *(const F4*)(A + (size_t)(bm + r) * Kd + (kk0 + lc4));
        U2 pk;
        pk.x = (unsigned)f2bf(v.x) | ((unsigned)f2bf(v.y) << 16);
        pk.y = (unsigned)f2bf(v.z) | ((unsigned)f2bf(v.w) << 16);
        *(U2*)&As[b][r][lc4] = pk;
      }
    } else {
#pragma unroll
      for (int i = 0; i < 16; i++) {
        int e = tid + (i << 7);
        int r = e >> 5, kk = e & 31;
        int gr = bm + r, gk = kk0 + kk;
        int cr = gr < M ? gr : M - 1;
        int ck = gk < Kd ? gk : Kd - 1;
        float v = A[(size_t)cr * Kd + ck];
        v = (gr < M && gk < Kd) ? v : 0.f;
        As[b][r][kk] = f2bf(v);
      }
    }
    // B tile 32x64, stored transposed
    if (vecB && fullK) {
#pragma unroll
      for (int p = 0; p < 4; p++) {
        int k = (tid >> 3) + ((p & 1) << 4);
        int cb = lc4 + ((p >> 1) << 5);
        F4 v = *(const F4*)(B + (size_t)(kk0 + k) * N + (bn + cb));
        Bt[b][cb + 0][k] = f2bf(v.x);
        Bt[b][cb + 1][k] = f2bf(v.y);
        Bt[b][cb + 2][k] = f2bf(v.z);
        Bt[b][cb + 3][k] = f2bf(v.w);
      }
    } else {
#pragma unroll
      for (int i = 0; i < 16; i++) {
        int e = tid + (i << 7);
        int k = e >> 6, c = e & 63;
        int gk = kk0 + k, gc = bn + c;
        int ck = gk < Kd ? gk : Kd - 1;
        int cc = gc < N ? gc : N - 1;
        float v = B[(size_t)ck * N + cc];
        v = (gk < Kd && gc < N) ? v : 0.f;
        Bt[b][c][k] = f2bf(v);
      }
    }
  };

  stage(0, 0);
  __syncthreads();
  int buf = 0;
  for (int k0 = 0; k0 < Kd; k0 += 32) {
    if (k0 + 32 < Kd) stage(buf ^ 1, k0 + 32);   // overlap with WMMA below
    // A fragments (rows wm*32.. +31): contiguous 8 bf16 per half -> b128
    int mr0 = (wm << 5) + ll;
    U8 r0, r1;
    r0.lo = *(const U4*)&As[buf][mr0][hh << 3];
    r0.hi = *(const U4*)&As[buf][mr0][16 + (hh << 3)];
    r1.lo = *(const U4*)&As[buf][mr0 + 16][hh << 3];
    r1.hi = *(const U4*)&As[buf][mr0 + 16][16 + (hh << 3)];
    v16bf a0 = __builtin_bit_cast(v16bf, r0);
    v16bf a1 = __builtin_bit_cast(v16bf, r1);
    // B fragments (cols wn*32.. +31): same K order as A (8 per half, two groups) -> b128
    int nc0 = (wn << 5) + ll;
    U8 s0, s1;
    s0.lo = *(const U4*)&Bt[buf][nc0][hh << 3];
    s0.hi = *(const U4*)&Bt[buf][nc0][16 + (hh << 3)];
    s1.lo = *(const U4*)&Bt[buf][nc0 + 16][hh << 3];
    s1.hi = *(const U4*)&Bt[buf][nc0 + 16][16 + (hh << 3)];
    v16bf b0 = __builtin_bit_cast(v16bf, s0);
    v16bf b1 = __builtin_bit_cast(v16bf, s1);

    acc00 = __builtin_amdgcn_wmma_f32_16x16x32_bf16(false, a0, false, b0, (short)0, acc00, false, false);
    acc01 = __builtin_amdgcn_wmma_f32_16x16x32_bf16(false, a0, false, b1, (short)0, acc01, false, false);
    acc10 = __builtin_amdgcn_wmma_f32_16x16x32_bf16(false, a1, false, b0, (short)0, acc10, false, false);
    acc11 = __builtin_amdgcn_wmma_f32_16x16x32_bf16(false, a1, false, b1, (short)0, acc11, false, false);
    __syncthreads();                              // single barrier per K-step
    buf ^= 1;
  }

  // D layout: VGPR j -> local row = j + 8*half
  int row0 = bm + (wm << 5) + (hh << 3);
  int col0 = bn + (wn << 5) + ll;
  int col1 = col0 + 16;
  int c0c = col0 < N ? col0 : N - 1;
  int c1c = col1 < N ? col1 : N - 1;
  float bv0 = bias ? bias[c0c] : 0.f;
  float bv1 = bias ? bias[c1c] : 0.f;
#pragma unroll
  for (int j = 0; j < 8; j++) {
    int ra = row0 + j;          // rows of acc0x
    int rb = ra + 16;           // rows of acc1x
    if (ra < M) {
      if (col0 < N) C[(size_t)ra * N + col0] = acc00[j] + bv0;
      if (col1 < N) C[(size_t)ra * N + col1] = acc01[j] + bv1;
    }
    if (rb < M) {
      if (col0 < N) C[(size_t)rb * N + col0] = acc10[j] + bv0;
      if (col1 < N) C[(size_t)rb * N + col1] = acc11[j] + bv1;
    }
  }
}

// ---------------------------------------------------------------------------
// GroupNorm(+activation) in place. One block per row; 8 waves = 8 groups,
// wave32 shuffle reduction. mode: 0=none, 1=leaky relu, 2=relu.
// ---------------------------------------------------------------------------
__global__ __launch_bounds__(256) void k_gn_act(
    float* __restrict__ x, const float* __restrict__ gg,
    const float* __restrict__ bb, int C, int mode) {
  int row = blockIdx.x;
  int wave = threadIdx.x >> 5;
  int lane = threadIdx.x & 31;
  int gs = C >> 3;                          // channels per group (GROUPS=8)
  float* xr = x + (size_t)row * C + (size_t)wave * gs;
  float s = 0.f, s2 = 0.f;
  for (int c = lane; c < gs; c += 32) { float v = xr[c]; s += v; s2 += v * v; }
#pragma unroll
  for (int o = 16; o > 0; o >>= 1) {
    s += __shfl_xor(s, o, 32);
    s2 += __shfl_xor(s2, o, 32);
  }
  float mean = s / (float)gs;
  float var = s2 / (float)gs - mean * mean;
  float rinv = rsqrtf(var + GN_EPS);
  for (int c = lane; c < gs; c += 32) {
    int ch = wave * gs + c;
    float v = (xr[c] - mean) * rinv * gg[ch] + bb[ch];
    if (mode == 1) v = v >= 0.f ? v : LRELU_NEG * v;
    else if (mode == 2) v = fmaxf(v, 0.f);
    xr[c] = v;
  }
}

// ---------------------------------------------------------------------------
// KPConv stage 1: agg[m,k,c] = sum_h infl(m,h,k) * feats[nbr[m,h], c].
// One block per query. Clamp+select gathers (branchless). Shadow idx==Ns -> 0.
// ---------------------------------------------------------------------------
__global__ __launch_bounds__(256) void k_gather_agg(
    const float* __restrict__ qp, const float* __restrict__ sp,
    const float* __restrict__ sf, const int* __restrict__ nb,
    const float* __restrict__ kp, float sigma, float* __restrict__ agg,
    int Ns, int C) {
  __shared__ float infl[HN][16];
  __shared__ float npt[HN][3];
  __shared__ int nid[HN];
  __shared__ float q[3];
  int t = threadIdx.x;
  long m = blockIdx.x;
  if (t < 3) q[t] = qp[m * 3 + t];
  if (t < HN) nid[t] = nb[m * HN + t];
  __syncthreads();
  if (t < HN * 3) {
    int h = t / 3, d = t - h * 3;
    int id = nid[h];
    int idc = id < Ns ? id : 0;
    float pv = sp[(size_t)idc * 3 + d];
    npt[h][d] = (id < Ns) ? pv : 1e6f;
  }
  __syncthreads();
  if (t < HN * KPTS) {
    int h = t / KPTS, k = t - h * KPTS;
    float dx = npt[h][0] - q[0] - kp[k * 3 + 0];
    float dy = npt[h][1] - q[1] - kp[k * 3 + 1];
    float dz = npt[h][2] - q[2] - kp[k * 3 + 2];
    float d = sqrtf(dx * dx + dy * dy + dz * dz + 1e-12f);
    infl[h][k] = fmaxf(1.f - d / sigma, 0.f);
  }
  __syncthreads();
  size_t base = (size_t)m * (size_t)(KPTS * C);
  for (int c = t; c < C; c += 256) {
    float av[KPTS];
#pragma unroll
    for (int k = 0; k < KPTS; k++) av[k] = 0.f;
#pragma unroll
    for (int h = 0; h < HN; h++) {
      int id = nid[h];
      int idc = id < Ns ? id : 0;
      float f = sf[(size_t)idc * C + c];
      f = (id < Ns) ? f : 0.f;
#pragma unroll
      for (int k = 0; k < KPTS; k++) av[k] += infl[h][k] * f;
    }
#pragma unroll
    for (int k = 0; k < KPTS; k++) agg[base + (size_t)k * C + c] = av[k];
  }
}

__global__ void k_maxpool(const float* __restrict__ f, const int* __restrict__ nb,
                          float* __restrict__ out, long total, int C, int Ns) {
  long i = (long)blockIdx.x * blockDim.x + threadIdx.x;
  if (i >= total) return;
  long m = i / C; int c = (int)(i - m * C);
  float mx = -1e10f;
#pragma unroll
  for (int h = 0; h < HN; h++) {
    int id = nb[m * HN + h];
    int idc = id < Ns ? id : 0;
    float v = f[(size_t)idc * C + c];
    mx = fmaxf(mx, (id < Ns) ? v : -1e10f);
  }
  out[i] = mx;
}

__global__ void k_add_lrelu(const float* __restrict__ a, const float* __restrict__ b,
                            float* __restrict__ o, long n) {
  long i = (long)blockIdx.x * blockDim.x + threadIdx.x;
  if (i >= n) return;
  float v = a[i] + b[i];
  o[i] = v >= 0.f ? v : LRELU_NEG * v;
}

__global__ void k_gather_cols(const float* __restrict__ src, const int* __restrict__ idx,
                              float* __restrict__ dst, int Cs, int Cdst, long total) {
  long i = (long)blockIdx.x * blockDim.x + threadIdx.x;
  if (i >= total) return;
  long m = i / Cs; int c = (int)(i - m * Cs);
  dst[m * (long)Cdst + c] = src[(size_t)idx[m] * Cs + c];
}

__global__ void k_copy_cols(const float* __restrict__ src, float* __restrict__ dst,
                            int Cs, int Cdst, long total) {
  long i = (long)blockIdx.x * blockDim.x + threadIdx.x;
  if (i >= total) return;
  long m = i / Cs; int c = (int)(i - m * Cs);
  dst[m * (long)Cdst + c] = src[m * (long)Cs + c];
}

__global__ void k_concat_ones(const float* __restrict__ f, float* __restrict__ o, long total) {
  long i = (long)blockIdx.x * blockDim.x + threadIdx.x;
  if (i >= total) return;
  long m = i / 5; int c = (int)(i - m * 5);
  o[i] = (c == 0) ? 1.f : f[m * 4 + (c - 1)];
}

// ---------------------------------------------------------------------------
// Host orchestration
// ---------------------------------------------------------------------------
struct LinP { const float *b, *be, *g, *w; };
struct ConvP { const float *be, *g, *kp, *w; };
struct ResP { ConvP cv; LinP sc; LinP u1, u2; bool hasSc; };

extern "C" void kernel_launch(void* const* d_in, const int* in_sizes, int n_in,
                              void* d_out, int out_size, void* d_ws, size_t ws_size,
                              hipStream_t stream) {
  (void)in_sizes; (void)n_in; (void)out_size; (void)ws_size;
  auto F = [&](int i) { return (const float*)d_in[i]; };
  auto I = [&](int i) { return (const int*)d_in[i]; };

  const float* feats = F(0);
  const float* pts[5] = { F(1), F(3), F(5), F(7), F(9) };
  const int* nbr[5] = { I(2), I(4), I(6), I(8), I(10) };
  const int* sub[4] = { I(11), I(13), I(15), I(17) };
  const int* up[4] = { I(12), I(14), I(16), I(18) };

  // params: jax pytree (sorted-key) leaf order starting at input 19.
  int pi = 19;
  auto nextLin = [&]() { LinP L{ F(pi), F(pi + 1), F(pi + 2), F(pi + 3) }; pi += 4; return L; };
  auto nextConv = [&]() { ConvP Cv{ F(pi), F(pi + 1), F(pi + 2), F(pi + 3) }; pi += 4; return Cv; };
  auto nextRes = [&](bool sc) {
    ResP R{}; R.cv = nextConv(); R.hasSc = sc;
    if (sc) R.sc = nextLin();
    R.u1 = nextLin(); R.u2 = nextLin(); return R;
  };
  LinP cls1 = nextLin();
  const float* cls2_b = F(pi); const float* cls2_w = F(pi + 1); pi += 2;
  LinP d1 = nextLin(), d2 = nextLin(), d3 = nextLin(), d4 = nextLin();
  ConvP e1_1 = nextConv();
  ResP e1_2 = nextRes(true);
  ResP e2_1 = nextRes(false), e2_2 = nextRes(true), e2_3 = nextRes(false);
  ResP e3_1 = nextRes(false), e3_2 = nextRes(true), e3_3 = nextRes(false);
  ResP e4_1 = nextRes(false), e4_2 = nextRes(true), e4_3 = nextRes(false);
  ResP e5_1 = nextRes(false), e5_2 = nextRes(true), e5_3 = nextRes(false);

  const float SG[5] = { 0.08f, 0.16f, 0.32f, 0.64f, 1.28f };

  // bump arena over d_ws
  char* wsb = (char*)d_ws;
  size_t off = 0;
  auto alloc = [&](size_t elems) {
    size_t b = (elems * sizeof(float) + 255) & ~(size_t)255;
    float* r = (float*)(wsb + off); off += b; return r;
  };

  auto gemm = [&](const float* A, const float* B, const float* bias, float* C,
                  int M, int Kd, int N) {
    dim3 g((unsigned)((N + 63) / 64), (unsigned)((M + 63) / 64));
    k_gemm<<<g, dim3(128), 0, stream>>>(A, B, bias, C, M, Kd, N);
  };
  auto gnact = [&](float* x, const float* g, const float* be, int M, int C, int mode) {
    k_gn_act<<<dim3((unsigned)M), dim3(256), 0, stream>>>(x, g, be, C, mode);
  };
  auto unary = [&](const float* x, int M, int Ci, const LinP& L, int Co, int mode, float* out) {
    gemm(x, L.w, L.b, out, M, Ci, Co);
    gnact(out, L.g, L.be, M, Co, mode);
  };
  auto conv_fwd = [&](const float* qp, const float* sp, const float* sfeat, const int* nb,
                      int Mq, int Ns, int Ci, int Co, const ConvP& cv, float sigma, float* out) {
    size_t mark = off;
    const int CH = 16384;
    int rows0 = Mq < CH ? Mq : CH;
    float* agg = alloc((size_t)rows0 * KPTS * Ci);
    for (int m0 = 0; m0 < Mq; m0 += CH) {
      int rows = (Mq - m0) < CH ? (Mq - m0) : CH;
      k_gather_agg<<<dim3((unsigned)rows), dim3(256), 0, stream>>>(
          qp + (size_t)m0 * 3, sp, sfeat, nb + (size_t)m0 * HN, cv.kp, sigma, agg, Ns, Ci);
      gemm(agg, cv.w, nullptr, out + (size_t)m0 * Co, rows, KPTS * Ci, Co);
    }
    gnact(out, cv.g, cv.be, Mq, Co, 1);
    off = mark;
  };
  auto resb = [&](const float* qp, const float* sp, const int* nb, int Mq, int Ns,
                  const float* fin, int ci, int co, const ResP& P, float sigma,
                  bool strided, float* fout) {
    size_t mark = off;
    int hc = co / 4;
    float* x1 = alloc((size_t)Ns * hc);
    unary(fin, Ns, ci, P.u1, hc, 1, x1);
    float* cx = alloc((size_t)Mq * hc);
    conv_fwd(qp, sp, x1, nb, Mq, Ns, hc, hc, P.cv, sigma, cx);
    float* x2 = alloc((size_t)Mq * co);
    unary(cx, Mq, hc, P.u2, co, 0, x2);
    const float* scv = fin;
    if (strided) {
      float* mp = alloc((size_t)Mq * ci);
      long tot = (long)Mq * ci;
      k_maxpool<<<dim3((unsigned)((tot + 255) / 256)), dim3(256), 0, stream>>>(
          fin, nb, mp, tot, ci, Ns);
      scv = mp;
    }
    if (P.hasSc) {
      float* so = alloc((size_t)Mq * co);
      unary(scv, Mq, ci, P.sc, co, 0, so);
      scv = so;
    }
    long tot = (long)Mq * co;
    k_add_lrelu<<<dim3((unsigned)((tot + 255) / 256)), dim3(256), 0, stream>>>(
        x2, scv, fout, tot);
    off = mark;
  };

  // persistent feature maps
  float* f1 = alloc((size_t)65536 * 128);
  float* f2 = alloc((size_t)16384 * 256);
  float* f3 = alloc((size_t)4096 * 512);
  float* f4 = alloc((size_t)1024 * 1024);
  float* f5 = alloc((size_t)256 * 2048);
  size_t mark0 = off;

  // ---- encoder ----
  {
    float* f1in = alloc((size_t)65536 * 5);
    long t = 65536L * 5;
    k_concat_ones<<<dim3((unsigned)((t + 255) / 256)), dim3(256), 0, stream>>>(feats, f1in, t);
    float* f1a = alloc((size_t)65536 * 64);
    conv_fwd(pts[0], pts[0], f1in, nbr[0], 65536, 65536, 5, 64, e1_1, SG[0], f1a);
    resb(pts[0], pts[0], nbr[0], 65536, 65536, f1a, 64, 128, e1_2, SG[0], false, f1);
    off = mark0;
  }
  resb(pts[1], pts[0], sub[0], 16384, 65536, f1, 128, 128, e2_1, SG[0], true, f2);
  resb(pts[1], pts[1], nbr[1], 16384, 16384, f2, 128, 256, e2_2, SG[1], false, f2);
  resb(pts[1], pts[1], nbr[1], 16384, 16384, f2, 256, 256, e2_3, SG[1], false, f2);
  resb(pts[2], pts[1], sub[1], 4096, 16384, f2, 256, 256, e3_1, SG[1], true, f3);
  resb(pts[2], pts[2], nbr[2], 4096, 4096, f3, 256, 512, e3_2, SG[2], false, f3);
  resb(pts[2], pts[2], nbr[2], 4096, 4096, f3, 512, 512, e3_3, SG[2], false, f3);
  resb(pts[3], pts[2], sub[2], 1024, 4096, f3, 512, 512, e4_1, SG[2], true, f4);
  resb(pts[3], pts[3], nbr[3], 1024, 1024, f4, 512, 1024, e4_2, SG[3], false, f4);
  resb(pts[3], pts[3], nbr[3], 1024, 1024, f4, 1024, 1024, e4_3, SG[3], false, f4);
  resb(pts[4], pts[3], sub[3], 256, 1024, f4, 1024, 1024, e5_1, SG[3], true, f5);
  resb(pts[4], pts[4], nbr[4], 256, 256, f5, 1024, 2048, e5_2, SG[4], false, f5);
  resb(pts[4], pts[4], nbr[4], 256, 256, f5, 2048, 2048, e5_3, SG[4], false, f5);

  // ---- decoder ----
  auto dec = [&](const float* hi, const int* upi, int M, int Chi,
                 const float* skip, int Csk, const LinP& L, int Co, float* out) {
    size_t mark = off;
    int Ct = Chi + Csk;
    float* cc = alloc((size_t)M * Ct);
    long t1 = (long)M * Chi;
    k_gather_cols<<<dim3((unsigned)((t1 + 255) / 256)), dim3(256), 0, stream>>>(
        hi, upi, cc, Chi, Ct, t1);
    long t2 = (long)M * Csk;
    k_copy_cols<<<dim3((unsigned)((t2 + 255) / 256)), dim3(256), 0, stream>>>(
        skip, cc + Chi, Csk, Ct, t2);
    unary(cc, M, Ct, L, Co, 1, out);
    off = mark;
  };

  float* l4 = alloc((size_t)1024 * 1024);
  dec(f5, up[3], 1024, 2048, f4, 1024, d4, 1024, l4);
  float* l3 = alloc((size_t)4096 * 512);
  dec(l4, up[2], 4096, 1024, f3, 512, d3, 512, l3);
  float* l2 = alloc((size_t)16384 * 256);
  dec(l3, up[1], 16384, 512, f2, 256, d2, 256, l2);
  float* l1 = alloc((size_t)65536 * 128);
  dec(l2, up[0], 65536, 256, f1, 128, d1, 128, l1);

  float* hb = alloc((size_t)65536 * 64);
  unary(l1, 65536, 128, cls1, 64, 2, hb);                   // GN + ReLU
  gemm(hb, cls2_w, cls2_b, (float*)d_out, 65536, 64, 13);   // logits
}
